// DLPTLayer_36550171688961
// MI455X (gfx1250) — compile-verified, matrix operand order checked by reading, not run on hardware
//
#include <hip/hip_runtime.h>

typedef __attribute__((ext_vector_type(16))) _Float16 v16h;
typedef __attribute__((ext_vector_type(8)))  float    v8f;

#define LDSS 40   // LDS tile row stride in halves (80 bytes -> multiple of 16B)

// ---------------------------------------------------------------------------
// Workhorse GEMM: 256 threads / 8 waves, 128x128 tile, double-buffered LDS.
// C[M,N] = alpha * A[M,K] @ B (+ bias[N])
//   A: f16 row-major (stride K). TRANSB=1: B is [N,K] row-major (weightsT /
//   K-matrix); TRANSB=0: B is [K,N] row-major.
// Each wave: 32 rows x 64 cols = 8x v_wmma_f32_16x16x32_f16 per K-step.
// Requires M%128==0, N%128==0, K%32==0.
// ---------------------------------------------------------------------------
template <int TRANSB>
__global__ __launch_bounds__(256) void gemm128_kernel(
    const _Float16* __restrict__ A, const _Float16* __restrict__ Bm,
    const float* __restrict__ bias,
    float* __restrict__ C32, _Float16* __restrict__ C16,
    int N, int K, long long sA, long long sB, long long sC, float alpha)
{
  __shared__ __align__(16) _Float16 lA[2][128 * LDSS];
  __shared__ __align__(16) _Float16 lB[2][128 * LDSS];   // n-major: lB[n][k]

  const int batch = blockIdx.z;
  A  += (long long)batch * sA;
  Bm += (long long)batch * sB;
  const long long cbase = (long long)batch * sC;

  const int m0   = blockIdx.y * 128;
  const int n0   = blockIdx.x * 128;
  const int tid  = threadIdx.x;
  const int lane = tid & 31;
  const int wave = tid >> 5;
  const int rowg = wave >> 1;          // 0..3 -> 32-row slice
  const int colg = wave & 1;           // 0..1 -> 64-col slice

  // loader coordinates
  const int ar = tid >> 1;             // 0..127 : A row in tile
  const int ac = (tid & 1) * 16;       // 0 / 16 : k-halves offset
  const _Float16* Arow = A + (long long)(m0 + ar) * K + ac;
  const _Float16* Brow = TRANSB ? (Bm + (long long)(n0 + ar) * K + ac) : nullptr;
  const int bn = tid & 127;            // non-trans: B col in tile
  const int bk = (tid >> 7) * 16;      // non-trans: k-halves offset

  v8f zero = {0.f,0.f,0.f,0.f,0.f,0.f,0.f,0.f};
  v8f acc[2][4];
#pragma unroll
  for (int r = 0; r < 2; ++r)
#pragma unroll
    for (int t = 0; t < 4; ++t) acc[r][t] = zero;

  uint4 sa0, sa1, sb0, sb1;
  _Float16 tb[16];

  // ---- prologue: fetch + store tile 0
  {
    const uint4* s4 = (const uint4*)(Arow);
    sa0 = s4[0]; sa1 = s4[1];
    if (TRANSB) {
      const uint4* b4 = (const uint4*)(Brow);
      sb0 = b4[0]; sb1 = b4[1];
    } else {
#pragma unroll
      for (int i = 0; i < 16; ++i)
        tb[i] = Bm[(long long)(bk + i) * N + n0 + bn];
    }
    uint4* d4 = (uint4*)&lA[0][ar * LDSS + ac];
    d4[0] = sa0; d4[1] = sa1;
    if (TRANSB) {
      uint4* e4 = (uint4*)&lB[0][ar * LDSS + ac];
      e4[0] = sb0; e4[1] = sb1;
    } else {
#pragma unroll
      for (int i = 0; i < 16; ++i) lB[0][bn * LDSS + bk + i] = tb[i];
    }
  }
  __syncthreads();

  int cur = 0;
  for (int k0 = 0; k0 < K; k0 += 32) {
    const bool more = (k0 + 32) < K;
    // ---- issue global fetches for next tile (land during the WMMAs)
    if (more) {
      const uint4* s4 = (const uint4*)(Arow + k0 + 32);
      __builtin_prefetch((const void*)(Arow + k0 + 64), 0, 1);
      sa0 = s4[0]; sa1 = s4[1];
      if (TRANSB) {
        const uint4* b4 = (const uint4*)(Brow + k0 + 32);
        sb0 = b4[0]; sb1 = b4[1];
      } else {
#pragma unroll
        for (int i = 0; i < 16; ++i)
          tb[i] = Bm[(long long)(k0 + 32 + bk + i) * N + n0 + bn];
      }
    }
    // ---- compute on current buffer: 2 A-frags x 4 B-frags = 8 WMMA
    {
      const int base = lane & 15;
      const int kg   = lane >> 4;
      union { v16h h; uint4 q[2]; } af[2];
#pragma unroll
      for (int r = 0; r < 2; ++r) {
        const int mrow = rowg * 32 + r * 16 + base;
        af[r].q[0] = *(const uint4*)&lA[cur][mrow * LDSS + kg * 8];
        af[r].q[1] = *(const uint4*)&lA[cur][mrow * LDSS + 16 + kg * 8];
      }
#pragma unroll
      for (int t = 0; t < 4; ++t) {
        const int nc = colg * 64 + t * 16 + base;
        union { v16h h; uint4 q[2]; } bf;
        bf.q[0] = *(const uint4*)&lB[cur][nc * LDSS + kg * 8];
        bf.q[1] = *(const uint4*)&lB[cur][nc * LDSS + 16 + kg * 8];
#pragma unroll
        for (int r = 0; r < 2; ++r)
          acc[r][t] = __builtin_amdgcn_wmma_f32_16x16x32_f16(
              false, af[r].h, false, bf.h, (short)0, acc[r][t], false, false);
      }
    }
    // ---- commit staged tile into the other buffer
    if (more) {
      uint4* d4 = (uint4*)&lA[cur ^ 1][ar * LDSS + ac];
      d4[0] = sa0; d4[1] = sa1;
      if (TRANSB) {
        uint4* e4 = (uint4*)&lB[cur ^ 1][ar * LDSS + ac];
        e4[0] = sb0; e4[1] = sb1;
      } else {
#pragma unroll
        for (int i = 0; i < 16; ++i) lB[cur ^ 1][bn * LDSS + bk + i] = tb[i];
      }
    }
    __syncthreads();
    cur ^= 1;
  }

  // ---- epilogue
  const int nlb = lane & 15;
  const int vb  = (lane >> 4) * 8;
#pragma unroll
  for (int r = 0; r < 2; ++r) {
    const int mbase = m0 + rowg * 32 + r * 16 + vb;
#pragma unroll
    for (int t = 0; t < 4; ++t) {
      const int n = n0 + colg * 64 + t * 16 + nlb;
      const float bv = bias ? bias[n] : 0.0f;
#pragma unroll
      for (int v = 0; v < 8; ++v) {
        const float val = acc[r][t][v] * alpha + bv;
        const long long off = cbase + (long long)(mbase + v) * N + n;
        if (C32) C32[off] = val;
        if (C16) C16[off] = (_Float16)val;
      }
    }
  }
}

// ---------------------------------------------------------------------------
// Small-tile GEMM (64x64, 128 threads) for the S=64 batched attention GEMMs.
// ---------------------------------------------------------------------------
template <int TRANSB>
__global__ __launch_bounds__(128) void gemm64_kernel(
    const _Float16* __restrict__ A, const _Float16* __restrict__ Bm,
    const float* __restrict__ bias,
    float* __restrict__ C32, _Float16* __restrict__ C16,
    int N, int K, long long sA, long long sB, long long sC, float alpha)
{
  __shared__ __align__(16) _Float16 lA[64 * LDSS];
  __shared__ __align__(16) _Float16 lB[64 * LDSS];

  const int batch = blockIdx.z;
  A  += (long long)batch * sA;
  Bm += (long long)batch * sB;
  const long long cbase = (long long)batch * sC;

  const int m0   = blockIdx.y * 64;
  const int n0   = blockIdx.x * 64;
  const int tid  = threadIdx.x;
  const int lane = tid & 31;
  const int wave = tid >> 5;

  v8f zero = {0.f,0.f,0.f,0.f,0.f,0.f,0.f,0.f};
  v8f acc[4];
#pragma unroll
  for (int t = 0; t < 4; ++t) acc[t] = zero;

  for (int k0 = 0; k0 < K; k0 += 32) {
    {
      int r = tid >> 1;
      int c = (tid & 1) * 16;
      const uint4* s4 = (const uint4*)(A + (long long)(m0 + r) * K + k0 + c);
      __builtin_prefetch((const void*)(A + (long long)(m0 + r) * K + k0 + 32), 0, 1);
      uint4 v0 = s4[0], v1 = s4[1];
      uint4* d4 = (uint4*)&lA[r * LDSS + c];
      d4[0] = v0; d4[1] = v1;
    }
    if (TRANSB) {
      int r = tid >> 1;
      int c = (tid & 1) * 16;
      const uint4* s4 = (const uint4*)(Bm + (long long)(n0 + r) * K + k0 + c);
      uint4 v0 = s4[0], v1 = s4[1];
      uint4* d4 = (uint4*)&lB[r * LDSS + c];
      d4[0] = v0; d4[1] = v1;
    } else {
      int n  = tid & 63;
      int kk = (tid >> 6) * 16;
#pragma unroll
      for (int i = 0; i < 16; ++i)
        lB[n * LDSS + kk + i] = Bm[(long long)(k0 + kk + i) * N + n0 + n];
    }
    __syncthreads();

    const int mrow = wave * 16 + (lane & 15);
    const int kg   = lane >> 4;
    union { v16h h; uint4 q[2]; } af;
    af.q[0] = *(const uint4*)&lA[mrow * LDSS + kg * 8];
    af.q[1] = *(const uint4*)&lA[mrow * LDSS + 16 + kg * 8];
#pragma unroll
    for (int t = 0; t < 4; ++t) {
      const int nc = t * 16 + (lane & 15);
      union { v16h h; uint4 q[2]; } bf;
      bf.q[0] = *(const uint4*)&lB[nc * LDSS + kg * 8];
      bf.q[1] = *(const uint4*)&lB[nc * LDSS + 16 + kg * 8];
      acc[t] = __builtin_amdgcn_wmma_f32_16x16x32_f16(
                 false, af.h, false, bf.h, (short)0, acc[t], false, false);
    }
    __syncthreads();
  }

  const int mbase = m0 + wave * 16 + (lane >> 4) * 8;
  const int nl    = lane & 15;
#pragma unroll
  for (int t = 0; t < 4; ++t) {
    const int n = n0 + t * 16 + nl;
    const float bv = bias ? bias[n] : 0.0f;
#pragma unroll
    for (int v = 0; v < 8; ++v) {
      const float val = acc[t][v] * alpha + bv;
      const long long off = cbase + (long long)(mbase + v) * N + n;
      if (C32) C32[off] = val;
      if (C16) C16[off] = (_Float16)val;
    }
  }
}

// ---------------------------------------------------------------------------
// Row-wise LayerNorm (+ optional residual add, optional ReLU), one wave/row.
// ---------------------------------------------------------------------------
template <int D>
__global__ __launch_bounds__(128) void ln_kernel(
    const float* __restrict__ X, const float* __restrict__ X2,
    const float* __restrict__ g, const float* __restrict__ be,
    float* __restrict__ Y32, _Float16* __restrict__ Y16, int relu)
{
  const long long row = (long long)blockIdx.x * 4 + (threadIdx.x >> 5);
  const int lane = threadIdx.x & 31;
  constexpr int PER = D / 32;
  float v[PER];
  const float* xr  = X + row * D;
  const float* x2r = X2 ? (X2 + row * D) : nullptr;
  float s = 0.f;
#pragma unroll
  for (int i = 0; i < PER; ++i) {
    int col = lane + i * 32;
    float t = xr[col];
    if (X2) t += x2r[col];
    v[i] = t; s += t;
  }
#pragma unroll
  for (int m = 16; m >= 1; m >>= 1) s += __shfl_xor(s, m);
  const float mean = s * (1.0f / D);
  float q = 0.f;
#pragma unroll
  for (int i = 0; i < PER; ++i) { float d0 = v[i] - mean; q += d0 * d0; }
#pragma unroll
  for (int m = 16; m >= 1; m >>= 1) q += __shfl_xor(q, m);
  const float istd = rsqrtf(q * (1.0f / D) + 1e-5f);
#pragma unroll
  for (int i = 0; i < PER; ++i) {
    int col = lane + i * 32;
    float y = (v[i] - mean) * istd * g[col] + be[col];
    if (relu) y = fmaxf(y, 0.0f);
    if (Y32) Y32[row * D + col] = y;
    if (Y16) Y16[row * D + col] = (_Float16)y;
  }
}

// ---------------------------------------------------------------------------
// Row-wise softmax over S keys, one wave per row, f16 output (feeds P@V GEMM)
// ---------------------------------------------------------------------------
template <int S>
__global__ __launch_bounds__(128) void softmax_kernel(
    const float* __restrict__ X, _Float16* __restrict__ P)
{
  const long long row = (long long)blockIdx.x * 4 + (threadIdx.x >> 5);
  const int lane = threadIdx.x & 31;
  constexpr int PER = S / 32;
  float v[PER];
  const float* xr = X + row * S;
  float mx = -1e30f;
#pragma unroll
  for (int i = 0; i < PER; ++i) { v[i] = xr[lane + i * 32]; mx = fmaxf(mx, v[i]); }
#pragma unroll
  for (int m = 16; m >= 1; m >>= 1) mx = fmaxf(mx, __shfl_xor(mx, m));
  float s = 0.f;
#pragma unroll
  for (int i = 0; i < PER; ++i) { v[i] = __expf(v[i] - mx); s += v[i]; }
#pragma unroll
  for (int m = 16; m >= 1; m >>= 1) s += __shfl_xor(s, m);
  const float inv = 1.0f / s;
#pragma unroll
  for (int i = 0; i < PER; ++i)
    P[row * S + lane + i * 32] = (_Float16)(v[i] * inv);
}

// ---------------------------------------------------------------------------
// Per-cluster prep: cog/local/nrm/avg + tiny MLPs 1a (4->64) and 2a (6->64),
// fused LN+ReLU, writes f16 concat rows A1 = [r || fc], A2 = [r_hat || fc].
// Cluster index is the identity partition, cluster c = rows [c*S, c*S+S).
// ---------------------------------------------------------------------------
__global__ void prep_kernel(
    const float* __restrict__ pos, const float* __restrict__ feat, int F,
    const float* __restrict__ W1a, const float* __restrict__ b1a,
    const float* __restrict__ g1a, const float* __restrict__ be1a,
    const float* __restrict__ W2a, const float* __restrict__ b2a,
    const float* __restrict__ g2a, const float* __restrict__ be2a,
    _Float16* __restrict__ A1, _Float16* __restrict__ A2, int S)
{
  const int Nn = 16384;
  const int c = blockIdx.x, b = blockIdx.y, s = threadIdx.x;
  const int point = c * S + s;
  const long long row = (long long)b * Nn + point;

  __shared__ float sb[256];
  auto bred = [&](float v) -> float {
    sb[s] = v; __syncthreads();
    for (int o = S >> 1; o > 0; o >>= 1) { if (s < o) sb[s] += sb[s + o]; __syncthreads(); }
    float r0 = sb[0]; __syncthreads(); return r0;
  };

  const float px = pos[row * 3 + 0], py = pos[row * 3 + 1], pz = pos[row * 3 + 2];
  const float invS = 1.0f / S;
  const float cx = bred(px) * invS, cy = bred(py) * invS, cz = bred(pz) * invS;
  const float lx = px - cx, ly = py - cy, lz = pz - cz;
  const float nr = sqrtf(lx * lx + ly * ly + lz * lz);
  const float ax = bred(lx) * invS, ay = bred(ly) * invS, az = bred(lz) * invS;

  const int KC = 64 + F;
  _Float16* a1r = A1 + row * KC;
  _Float16* a2r = A2 + row * KC;

  float r[64];
#pragma unroll
  for (int j = 0; j < 64; ++j)
    r[j] = lx * W1a[j] + ly * W1a[64 + j] + lz * W1a[128 + j] + nr * W1a[192 + j] + b1a[j];
  {
    float m = 0.f;
#pragma unroll
    for (int j = 0; j < 64; ++j) m += r[j];
    m *= (1.0f / 64.0f);
    float v = 0.f;
#pragma unroll
    for (int j = 0; j < 64; ++j) { float d0 = r[j] - m; v += d0 * d0; }
    const float is = rsqrtf(v * (1.0f / 64.0f) + 1e-5f);
#pragma unroll
    for (int j = 0; j < 64; ++j) {
      float y = (r[j] - m) * is * g1a[j] + be1a[j];
      a1r[j] = (_Float16)fmaxf(y, 0.0f);
    }
  }
#pragma unroll
  for (int j = 0; j < 64; ++j)
    r[j] = ax * W2a[j] + ay * W2a[64 + j] + az * W2a[128 + j] +
           lx * W2a[192 + j] + ly * W2a[256 + j] + lz * W2a[320 + j] + b2a[j];
  {
    float m = 0.f;
#pragma unroll
    for (int j = 0; j < 64; ++j) m += r[j];
    m *= (1.0f / 64.0f);
    float v = 0.f;
#pragma unroll
    for (int j = 0; j < 64; ++j) { float d0 = r[j] - m; v += d0 * d0; }
    const float is = rsqrtf(v * (1.0f / 64.0f) + 1e-5f);
#pragma unroll
    for (int j = 0; j < 64; ++j) {
      float y = (r[j] - m) * is * g2a[j] + be2a[j];
      a2r[j] = (_Float16)fmaxf(y, 0.0f);
    }
  }
  const float* fr = feat + row * F;
  for (int j = 0; j < F; ++j) {
    _Float16 h = (_Float16)fr[j];
    a1r[64 + j] = h;
    a2r[64 + j] = h;
  }
}

// ---------------------------------------------------------------------------
// Weight convert + transpose: w[K,N] f32 -> wT[N,K] f16
// ---------------------------------------------------------------------------
__global__ void f32_to_f16t_kernel(const float* __restrict__ w,
                                   _Float16* __restrict__ wt, int K, int N)
{
  long long i = (long long)blockIdx.x * blockDim.x + threadIdx.x;
  if (i >= (long long)K * N) return;
  int n = (int)(i / K);
  int k = (int)(i % K);
  wt[i] = (_Float16)w[(long long)k * N + n];
}

__global__ void gather_pos_kernel(const float* __restrict__ pos,
                                  const int* __restrict__ fps,
                                  float* __restrict__ out, int N, int NDS)
{
  int i = blockIdx.x * 256 + threadIdx.x;       // over B*NDS
  if (i >= 4 * NDS) return;
  int b = i / NDS;
  int idx = fps[i];
  long long src = ((long long)b * N + idx) * 3;
#pragma unroll
  for (int k = 0; k < 3; ++k) out[(long long)i * 3 + k] = pos[src + k];
}

__global__ void gather_feat_kernel(const float* __restrict__ feat,
                                   const int* __restrict__ fps,
                                   float* __restrict__ out, int N, int NDS, int D)
{
  long long i = (long long)blockIdx.x * 256 + threadIdx.x;   // over B*NDS*D
  if (i >= 4LL * NDS * D) return;
  long long row = i / D;
  int col = (int)(i % D);
  int b = (int)(row / NDS);
  int idx = fps[row];
  out[i] = feat[((long long)b * N + idx) * (long long)D + col];
}

// ---------------------------------------------------------------------------
extern "C" void kernel_launch(void* const* d_in, const int* in_sizes, int n_in,
                              void* d_out, int out_size, void* d_ws, size_t ws_size,
                              hipStream_t stream)
{
  (void)in_sizes; (void)n_in; (void)out_size; (void)ws_size;
  const int Bb = 4, Nn = 16384, NDS = Nn / 4;
  const long long BN = (long long)Bb * Nn;

  const float* pos   = (const float*)d_in[0];
  const float* feat0 = (const float*)d_in[1];
  const int*   fps   = (const int*)d_in[4];
  auto prm = [&](int idx) -> const float* { return (const float*)d_in[idx]; };

  // ---- deterministic workspace carve-out -------------------------------
  char* ws = (char*)d_ws;
  size_t off = 0;
  auto alloc = [&](size_t bytes) -> void* {
    off = (off + 255) & ~(size_t)255;
    void* p = ws + off;
    off += bytes;
    return p;
  };
  _Float16* w1b[2]; _Float16* w2b[2];
  _Float16* wq[2];  _Float16* wk[2];  _Float16* wv[2];  _Float16* wo[2];
  w1b[0] = (_Float16*)alloc(128 * 256 * 2); w2b[0] = (_Float16*)alloc(128 * 256 * 2);
  wq[0] = (_Float16*)alloc(256 * 256 * 2);  wk[0] = (_Float16*)alloc(256 * 256 * 2);
  wv[0] = (_Float16*)alloc(256 * 256 * 2);  wo[0] = (_Float16*)alloc(256 * 256 * 2);
  w1b[1] = (_Float16*)alloc(320 * 512 * 2); w2b[1] = (_Float16*)alloc(320 * 512 * 2);
  wq[1] = (_Float16*)alloc(512 * 512 * 2);  wk[1] = (_Float16*)alloc(512 * 512 * 2);
  wv[1] = (_Float16*)alloc(512 * 512 * 2);  wo[1] = (_Float16*)alloc(512 * 512 * 2);
  _Float16* A1h    = (_Float16*)alloc(BN * 320 * 2);
  _Float16* A2h    = (_Float16*)alloc(BN * 320 * 2);
  float*    T32    = (float*)   alloc(BN * 512 * 4);
  float*    hpos32 = (float*)   alloc(BN * 512 * 4);
  _Float16* hposh  = (_Float16*)alloc(BN * 512 * 2);
  _Float16* hgeoh  = (_Float16*)alloc(BN * 512 * 2);
  _Float16* Qh     = (_Float16*)alloc(BN * 512 * 2);
  _Float16* Kh     = (_Float16*)alloc(BN * 512 * 2);
  _Float16* Vh     = (_Float16*)alloc(BN * 512 * 2);
  float*    Sc32   = (float*)   alloc((long long)256 * 65536 * 4);  // 256 clusters * 256*256
  _Float16* Ph     = (_Float16*)alloc((long long)256 * 65536 * 2);
  _Float16* AVh    = (_Float16*)alloc(BN * 512 * 2);
  float*    featb  = (float*)   alloc(BN * 512 * 4);

  // generic GEMM dispatch: picks 128x128 double-buffered kernel when legal
  auto gemm = [&](const _Float16* A, const _Float16* Bmat, const float* bias,
                  float* C32, _Float16* C16, int M, int Nc, int K,
                  long long sA, long long sB, long long sC, int batch,
                  int transB, float alpha) {
    if ((M % 128 == 0) && (Nc % 128 == 0)) {
      dim3 g(Nc / 128, M / 128, batch);
      if (transB)
        gemm128_kernel<1><<<g, 256, 0, stream>>>(A, Bmat, bias, C32, C16, Nc, K, sA, sB, sC, alpha);
      else
        gemm128_kernel<0><<<g, 256, 0, stream>>>(A, Bmat, bias, C32, C16, Nc, K, sA, sB, sC, alpha);
    } else {
      dim3 g(Nc / 64, M / 64, batch);
      if (transB)
        gemm64_kernel<1><<<g, 128, 0, stream>>>(A, Bmat, bias, C32, C16, Nc, K, sA, sB, sC, alpha);
      else
        gemm64_kernel<0><<<g, 128, 0, stream>>>(A, Bmat, bias, C32, C16, Nc, K, sA, sB, sC, alpha);
    }
  };

  struct Blk { int F, d, S, C, Kc, pb; float scale; };
  const Blk bp[2] = {
    { 64,  256,  64, 256, 128,  5, 1.0f / 16.0f },
    { 256, 512, 256,  64, 320, 28, 0.04419417382f },  // 1/sqrt(512)
  };

  const float* featIn = feat0;
  for (int blk = 0; blk < 2; ++blk) {
    const Blk& p = bp[blk];
    const int pb = p.pb, d = p.d, S = p.S, C = p.C, Kc = p.Kc, F = p.F;
    const float scale = p.scale;

    // weight conversion + transpose (f32 [K,N] -> f16 [N,K])
    auto cvt = [&](int pi, _Float16* dst, int K, int Nc) {
      long long n = (long long)K * Nc;
      f32_to_f16t_kernel<<<(int)((n + 255) / 256), 256, 0, stream>>>(prm(pi), dst, K, Nc);
    };
    cvt(pb + 4,  w1b[blk], Kc, d);
    cvt(pb + 12, w2b[blk], Kc, d);
    cvt(pb + 16, wq[blk],  d, d);
    cvt(pb + 17, wk[blk],  d, d);
    cvt(pb + 18, wv[blk],  d, d);
    cvt(pb + 19, wo[blk],  d, d);

    prep_kernel<<<dim3(C, Bb), S, 0, stream>>>(
        pos, featIn, F,
        prm(pb + 0), prm(pb + 1), prm(pb + 2), prm(pb + 3),
        prm(pb + 8), prm(pb + 9), prm(pb + 10), prm(pb + 11),
        A1h, A2h, S);

    auto ln = [&](const float* X, const float* X2, const float* g, const float* be,
                  float* Y32, _Float16* Y16, int relu) {
      if (d == 256)
        ln_kernel<256><<<(int)(BN / 4), 128, 0, stream>>>(X, X2, g, be, Y32, Y16, relu);
      else
        ln_kernel<512><<<(int)(BN / 4), 128, 0, stream>>>(X, X2, g, be, Y32, Y16, relu);
    };

    // h_pos = relu(LN(A1 @ W1b + b1b))
    gemm(A1h, w1b[blk], prm(pb + 5), T32, nullptr, (int)BN, d, Kc, 0, 0, 0, 1, 1, 1.0f);
    ln(T32, nullptr, prm(pb + 6), prm(pb + 7), hpos32, hposh, 1);

    // h_geo = relu(LN(A2 @ W2b + b2b))
    gemm(A2h, w2b[blk], prm(pb + 13), T32, nullptr, (int)BN, d, Kc, 0, 0, 0, 1, 1, 1.0f);
    ln(T32, nullptr, prm(pb + 14), prm(pb + 15), nullptr, hgeoh, 1);

    // Q,K from h_geo; V from h_pos
    gemm(hgeoh, wq[blk], nullptr, nullptr, Qh, (int)BN, d, d, 0, 0, 0, 1, 1, 1.0f);
    gemm(hgeoh, wk[blk], nullptr, nullptr, Kh, (int)BN, d, d, 0, 0, 0, 1, 1, 1.0f);
    gemm(hposh, wv[blk], nullptr, nullptr, Vh, (int)BN, d, d, 0, 0, 0, 1, 1, 1.0f);

    // per-cluster scores = scale * Q @ K^T  (batched, transB)
    const int batch = Bb * C;
    gemm(Qh, Kh, nullptr, Sc32, nullptr, S, S, d,
         (long long)S * d, (long long)S * d, (long long)S * S, batch, 1, scale);

    if (S == 64)
      softmax_kernel<64><<<(int)(BN / 4), 128, 0, stream>>>(Sc32, Ph);
    else
      softmax_kernel<256><<<(int)(BN / 4), 128, 0, stream>>>(Sc32, Ph);

    // attn = P @ V (batched, B not transposed)
    gemm(Ph, Vh, nullptr, nullptr, AVh, S, d, S,
         (long long)S * S, (long long)S * d, (long long)S * d, batch, 0, 1.0f);

    // O = attn @ Wo + bo
    gemm(AVh, wo[blk], prm(pb + 20), T32, nullptr, (int)BN, d, d, 0, 0, 0, 1, 1, 1.0f);

    // feat_out = LN(h_pos + O)
    ln(hpos32, T32, prm(pb + 21), prm(pb + 22), featb, nullptr, 0);

    featIn = featb;
  }

  // ---- final FPS gather: d_out = [pos_ds (B*NDS*3) | feat_ds (B*NDS*512)]
  float* out = (float*)d_out;
  float* out_pos  = out;
  float* out_feat = out + (long long)Bb * NDS * 3;
  gather_pos_kernel<<<(Bb * NDS + 255) / 256, 256, 0, stream>>>(pos, fps, out_pos, Nn, NDS);
  {
    long long n = (long long)Bb * NDS * 512;
    gather_feat_kernel<<<(int)((n + 255) / 256), 256, 0, stream>>>(featb, fps, out_feat, Nn, NDS, 512);
  }
}
